// RNN_41936060678228
// MI455X (gfx1250) — matrix operand order repaired and checked by reading, not compiled, not run on hardware
//
#include <hip/hip_runtime.h>

// CDNA5 / gfx1250, wave32. Sequential LSTM log-psi.
// Design: 128 blocks x 128 threads (4 waves). Each block owns 16 batch rows,
// iterates the 513 time steps with h/c resident; per-step GEMM [16,64]@[64,256]
// via v_wmma_f32_16x16x32_f16 (f16 inputs, f32 accum). Wh fragments + gate
// biases in registers; h exchanged via double-buffered LDS f16 tile.

typedef _Float16 v16h __attribute__((ext_vector_type(16)));
typedef _Float16 v8h  __attribute__((ext_vector_type(8)));
typedef float    v8f  __attribute__((ext_vector_type(8)));
typedef int      v4i  __attribute__((ext_vector_type(4)));

#define L_SEQ 512
#define F_DIM 64
#define G4    256   // 4*F

__device__ __forceinline__ float sigf(float x)   { return 1.0f / (1.0f + __expf(-x)); }
__device__ __forceinline__ float tanhf_(float x) { return 1.0f - 2.0f / (__expf(2.0f * x) + 1.0f); }

__global__ __launch_bounds__(128)
void RNN_logpsi_kernel(const int* __restrict__ s,
                       const float* __restrict__ W0, const float* __restrict__ b0,
                       const float* __restrict__ Wi, const float* __restrict__ Wh,
                       const float* __restrict__ bh, const float* __restrict__ Wa,
                       const float* __restrict__ ba, const float* __restrict__ Wp,
                       const float* __restrict__ bp, float* __restrict__ out)
{
    __shared__ int sbufT[L_SEQ][16];                       // spins, transposed [t][m]
    __shared__ __align__(16) _Float16 hbuf[2][16][72];     // padded rows (144B)

    const int tid   = threadIdx.x;
    const int lane  = tid & 31;
    const int wv    = tid >> 5;          // wave 0..3 -> feature group
    const int rowB  = blockIdx.x * 16;   // batch tile base

    // ---- stage spins: coalesced global read, transposed LDS write ----
    for (int idx = tid; idx < 16 * L_SEQ; idx += 128) {
        int r = idx >> 9;            // /512
        int c = idx & 511;
        sbufT[c][r] = s[(rowB + r) * L_SEQ + c];
    }
    // ---- zero h buffers (h0 = 0) ----
    for (int idx = tid; idx < 2 * 16 * 72; idx += 128)
        ((_Float16*)hbuf)[idx] = (_Float16)0.0f;

    // ---- per-lane geometry ----
    const int n15   = lane & 15;
    const int mBase = (lane >> 4) << 3;      // 0 or 8
    const int feat  = wv * 16 + n15;         // feature owned by this lane
    const int hiOff = (lane >> 4) * 8;

    // ---- Wh B-fragments (f16) kept in registers: [gate][Kchunk] ----
    v16h bf[4][2];
    #pragma unroll
    for (int g = 0; g < 4; ++g) {
        int ncol = g * 64 + feat;
        #pragma unroll
        for (int kc = 0; kc < 2; ++kc) {
            int kb = kc * 32 + ((lane >> 4) << 3);
            #pragma unroll
            for (int j = 0; j < 16; ++j)
                bf[g][kc][j] = (_Float16)Wh[(kb + j + ((j >> 3) << 3)) * G4 + ncol];
        }
    }

    // ---- gate biases per lane: G0 (spin 0) and delta (spin 1 - spin 0) ----
    float g0r[4], gdr[4];
    #pragma unroll
    for (int g = 0; g < 4; ++g) {
        int col = g * 64 + feat;
        float a0 = bh[col], a1 = bh[col];
        for (int f = 0; f < F_DIM; ++f) {
            float w = Wi[f * G4 + col];
            a0 += (W0[f]         + b0[f]) * w;
            a1 += (W0[F_DIM + f] + b0[f]) * w;
        }
        g0r[g] = a0;
        gdr[g] = a1 - a0;
    }

    // ---- logits thread mapping: 16 rows x (2 classes x 4 chunks) = 128 ----
    const int lm  = tid >> 3;
    const int lk  = tid & 1;
    const int lch = (tid >> 1) & 3;
    float waR[16];
    #pragma unroll
    for (int j = 0; j < 16; ++j) waR[j] = Wa[(lch * 16 + j) * 2 + lk];
    const float ba0 = ba[0], ba1 = ba[1];

    v8f creg;
    #pragma unroll
    for (int r = 0; r < 8; ++r) creg[r] = 0.0f;
    float amp = 0.0f;

    __syncthreads();

    // =========================== time loop ===========================
    for (int t = 0; t <= L_SEQ; ++t) {
        const int cb = t & 1, nb = cb ^ 1;

        // A fragments of h(prev): 16x64 f16, two K32 chunks
        const _Float16* hrow = &hbuf[cb][n15][0];
        v8h x0 = *(const v8h*)(hrow + hiOff);
        v8h x1 = *(const v8h*)(hrow + 16 + hiOff);
        v8h x2 = *(const v8h*)(hrow + 32 + hiOff);
        v8h x3 = *(const v8h*)(hrow + 48 + hiOff);
        v16h a0, a1;
        #pragma unroll
        for (int j = 0; j < 8; ++j) {
            a0[j] = x0[j]; a0[8 + j] = x1[j];
            a1[j] = x2[j]; a1[8 + j] = x3[j];
        }

        // 4 gates x 2 K-chunks of v_wmma_f32_16x16x32_f16
        v8f acc[4];
        #pragma unroll
        for (int g = 0; g < 4; ++g) {
            v8f z;
            #pragma unroll
            for (int r = 0; r < 8; ++r) z[r] = 0.0f;
            z = __builtin_amdgcn_wmma_f32_16x16x32_f16(false, a0, false, bf[g][0],
                                                       (short)0, z, false, false);
            z = __builtin_amdgcn_wmma_f32_16x16x32_f16(false, a1, false, bf[g][1],
                                                       (short)0, z, false, false);
            acc[g] = z;
        }

        // input spin per batch row (s_pad: step 0 is zero token)
        float spf[8];
        if (t == 0) {
            #pragma unroll
            for (int r = 0; r < 8; ++r) spf[r] = 0.0f;
        } else {
            v4i sa = *(const v4i*)(&sbufT[t - 1][mBase]);
            v4i sb = *(const v4i*)(&sbufT[t - 1][mBase + 4]);
            spf[0] = (float)sa[0]; spf[1] = (float)sa[1];
            spf[2] = (float)sa[2]; spf[3] = (float)sa[3];
            spf[4] = (float)sb[0]; spf[5] = (float)sb[1];
            spf[6] = (float)sb[2]; spf[7] = (float)sb[3];
        }

        // LSTM cell nonlinearity; c stays in registers, h -> LDS f16
        #pragma unroll
        for (int r = 0; r < 8; ++r) {
            float iv = acc[0][r] + g0r[0] + spf[r] * gdr[0];
            float fv = acc[1][r] + g0r[1] + spf[r] * gdr[1];
            float gv = acc[2][r] + g0r[2] + spf[r] * gdr[2];
            float ov = acc[3][r] + g0r[3] + spf[r] * gdr[3];
            float cn = sigf(fv) * creg[r] + sigf(iv) * tanhf_(gv);
            float hn = sigf(ov) * tanhf_(cn);
            creg[r] = cn;
            hbuf[nb][mBase + r][feat] = (_Float16)hn;
        }
        __syncthreads();

        // logits/log-softmax (K=2) on ys[t] for t < L
        if (t < L_SEQ) {
            const _Float16* hr = &hbuf[nb][lm][lch * 16];
            float p = 0.0f;
            #pragma unroll
            for (int j = 0; j < 16; ++j) p += (float)hr[j] * waR[j];
            p += __shfl_xor(p, 2, 32);              // reduce chunks
            p += __shfl_xor(p, 4, 32);
            float other = __shfl_xor(p, 1, 32);     // other class
            float c0v = (lk == 0) ? p : other;
            float c1v = (lk == 0) ? other : p;
            float l0 = c0v + ba0, l1 = c1v + ba1;
            float mx  = fmaxf(l0, l1);
            float lse = mx + __logf(__expf(l0 - mx) + __expf(l1 - mx));
            int   sp  = sbufT[t][lm];
            float pick = sp ? l1 : l0;
            if ((tid & 7) == 0) amp += 0.5f * (pick - lse);
        }
    }

    // ---- outputs: complex64 interleaved (amp, phase) per batch row ----
    if ((tid & 7) == 0) out[2 * (rowB + (tid >> 3))] = amp;
    if (tid < 16) {
        float ph = bp[0];
        const _Float16* hr = &hbuf[1][tid][0];   // h after step L (L+1 steps, buf (L+1)&1)
        #pragma unroll
        for (int f = 0; f < F_DIM; ++f) ph += (float)hr[f] * Wp[f];
        out[2 * (rowB + tid) + 1] = ph;
    }
}

extern "C" void kernel_launch(void* const* d_in, const int* in_sizes, int n_in,
                              void* d_out, int out_size, void* d_ws, size_t ws_size,
                              hipStream_t stream) {
    (void)in_sizes; (void)n_in; (void)out_size; (void)d_ws; (void)ws_size;
    const int*   s_  = (const int*)  d_in[0];
    const float* W0  = (const float*)d_in[1];
    const float* b0_ = (const float*)d_in[2];
    const float* Wi  = (const float*)d_in[3];
    const float* Wh  = (const float*)d_in[4];
    const float* bh  = (const float*)d_in[5];
    const float* Wa  = (const float*)d_in[6];
    const float* ba  = (const float*)d_in[7];
    const float* Wp  = (const float*)d_in[8];
    const float* bp  = (const float*)d_in[9];
    float* out = (float*)d_out;

    RNN_logpsi_kernel<<<2048 / 16, 128, 0, stream>>>(s_, W0, b0_, Wi, Wh, bh,
                                                     Wa, ba, Wp, bp, out);
}
